// GCN_4226247819850
// MI455X (gfx1250) — matrix operand order repaired and checked by reading, not compiled, not run on hardware
//
#include <hip/hip_runtime.h>
#include <hip/hip_bf16.h>

typedef __attribute__((ext_vector_type(16))) _Float16 v16h;
typedef __attribute__((ext_vector_type(8)))  float    v8f;

#define HF 64   // hidden/feature width
#define NGRAPH 64
#define NCLASS 10

// ---------------- degree / normalization ----------------
__global__ void deg_init_kernel(float* deg, int n) {
    int i = blockIdx.x * blockDim.x + threadIdx.x;
    if (i < n) deg[i] = 1.0f;              // self-loop contributes 1
}

__global__ void deg_edges_kernel(const int* __restrict__ tgt, float* deg, int e) {
    int i = blockIdx.x * blockDim.x + threadIdx.x;
    if (i < e) atomicAdd(&deg[tgt[i]], 1.0f);
}

__global__ void deg_rsqrt_kernel(float* deg, int n) {
    int i = blockIdx.x * blockDim.x + threadIdx.x;
    if (i < n) deg[i] = rsqrtf(deg[i]);    // deg >= 1 always (self loops)
}

// ---------------- WMMA GEMM: H[n,64] = act(X[n,64]) @ W[64,64] ----------------
// One wave computes a 16x64 output strip: 4 N-tiles x 2 K-steps of
// v_wmma_f32_16x16x32_f16. 256-thread block = 8 waves = 128 rows.
// W pre-swizzled into LDS in fragment order -> 2x ds_load_b128 per fragment.
// H must have one pad row at index n (dump target for OOB lanes); stores are
// unconditional with a cndmask'd row index -> no EXEC juggling.
__global__ void gcn_gemm_wmma_kernel(const float* __restrict__ X,
                                     const float* __restrict__ W,
                                     float* __restrict__ H,
                                     int n, int apply_relu) {
    __shared__ _Float16 Wlds[HF * HF];     // 8 KB, fragment-ordered
    const int tid = threadIdx.x;
    // swizzled fill: element (k,col) -> frag (t=col/16, ks=k/32),
    // lane = (k%32)/16*16 + col%16, slot j = k%16
    for (int idx = tid; idx < HF * HF; idx += blockDim.x) {
        const int k = idx >> 6, col = idx & 63;
        const int t = col >> 4, l16c = col & 15;
        const int ks = k >> 5, kk = k & 31;
        const int hic = kk >> 4, j = kk & 15;
        const int pos = (((t * 2 + ks) * 32) + hic * 16 + l16c) * 16 + j;
        Wlds[pos] = (_Float16)W[idx];
    }
    __syncthreads();

    const int wave = tid >> 5;
    const int lane = tid & 31;
    const int l16  = lane & 15;
    const int hi   = lane >> 4;
    const int rowBase = (blockIdx.x * 8 + wave) * 16;
    if (rowBase >= n) return;              // wave-uniform: EXEC stays all-1s

    // B fragments: contiguous 32B vector load per lane (2x ds_load_b128)
    v16h bfrag[4][2];
    #pragma unroll
    for (int t = 0; t < 4; ++t)
        #pragma unroll
        for (int ks = 0; ks < 2; ++ks)
            bfrag[t][ks] = *(const v16h*)(Wlds + (((t * 2 + ks) * 32) + lane) * 16);

    // A fragments: float4 global loads, f32->f16 convert, optional fused ReLU
    const int row = rowBase + l16;
    const float* xr = X + (size_t)((row < n) ? row : (n - 1)) * HF;
    v16h afrag[2];
    #pragma unroll
    for (int ks = 0; ks < 2; ++ks) {
        const int kb0 = ks * 32 + hi * 16;       // elems 0-7
        const int kb1 = ks * 32 + hi * 16 + 8;   // elems 8-15
        const float4 p0 = *(const float4*)(xr + kb0);
        const float4 p1 = *(const float4*)(xr + kb0 + 4);
        const float4 p2 = *(const float4*)(xr + kb1);
        const float4 p3 = *(const float4*)(xr + kb1 + 4);
        float v[16] = { p0.x,p0.y,p0.z,p0.w, p1.x,p1.y,p1.z,p1.w,
                        p2.x,p2.y,p2.z,p2.w, p3.x,p3.y,p3.z,p3.w };
        v16h a;
        #pragma unroll
        for (int j = 0; j < 16; ++j) {
            float f = v[j];
            if (apply_relu) f = fmaxf(f, 0.0f);
            a[j] = (_Float16)f;
        }
        afrag[ks] = a;
    }

    #pragma unroll
    for (int t = 0; t < 4; ++t) {
        v8f c = {0.f,0.f,0.f,0.f,0.f,0.f,0.f,0.f};
        c = __builtin_amdgcn_wmma_f32_16x16x32_f16(false, afrag[0], false, bfrag[t][0],
                                                   (short)0, c, false, false);
        c = __builtin_amdgcn_wmma_f32_16x16x32_f16(false, afrag[1], false, bfrag[t][1],
                                                   (short)0, c, false, false);
        // C/D layout: elem r -> row hi*8+r, col l16; OOB lanes -> pad row n
        #pragma unroll
        for (int r = 0; r < 8; ++r) {
            int orow = rowBase + hi * 8 + r;
            orow = (orow < n) ? orow : n;          // cndmask, no branch
            H[(size_t)orow * HF + t * 16 + l16] = c[r];
        }
    }
}

// ---------------- aggregation ----------------
// out[i, 4f..4f+3] = dis[i]^2 * h[i, ...] + bias[...]   (self-loop + bias)
__global__ void agg_init_kernel(const float* __restrict__ h,
                                const float* __restrict__ dis,
                                const float* __restrict__ bias,
                                float* __restrict__ out, int n) {
    long long idx = (long long)blockIdx.x * blockDim.x + threadIdx.x;
    if (idx < (long long)n * 16) {
        const int i = (int)(idx >> 4), q = ((int)idx & 15) * 4;
        const float d = dis[i], dd = d * d;
        const float4 hv = *(const float4*)(h + (size_t)i * HF + q);
        const float4 bv = *(const float4*)(bias + q);
        float4 r;
        r.x = dd * hv.x + bv.x;  r.y = dd * hv.y + bv.y;
        r.z = dd * hv.z + bv.z;  r.w = dd * hv.w + bv.w;
        *(float4*)(out + (size_t)i * HF + q) = r;
    }
}

// out[tgt, 4f..] += dis[src]*dis[tgt]*h[src, 4f..]  (one lane per edge-quad)
__global__ void agg_edges_kernel(const int* __restrict__ src,
                                 const int* __restrict__ tgt,
                                 const float* __restrict__ h,
                                 const float* __restrict__ dis,
                                 float* __restrict__ out, int e) {
    long long idx = (long long)blockIdx.x * blockDim.x + threadIdx.x;
    if (idx < (long long)e * 16) {
        const int ed = (int)(idx >> 4), q = ((int)idx & 15) * 4;
        const int s = src[ed], t = tgt[ed];
        const float norm = dis[s] * dis[t];
        const float4 hv = *(const float4*)(h + (size_t)s * HF + q);
        float* o = out + (size_t)t * HF + q;
        atomicAdd(o + 0, norm * hv.x);
        atomicAdd(o + 1, norm * hv.y);
        atomicAdd(o + 2, norm * hv.z);
        atomicAdd(o + 3, norm * hv.w);
    }
}

// ---------------- pooling + classifier ----------------
__global__ void pool_zero_kernel(float* pooled, float* cnt) {
    int i = blockIdx.x * blockDim.x + threadIdx.x;
    if (i < NGRAPH * HF) pooled[i] = 0.0f;
    if (i < NGRAPH) cnt[i] = 0.0f;
}

__global__ void pool_sum_kernel(const float* __restrict__ act,
                                const int* __restrict__ batch,
                                float* pooled, float* cnt, int n) {
    long long idx = (long long)blockIdx.x * blockDim.x + threadIdx.x;
    if (idx < (long long)n * 16) {
        const int i = (int)(idx >> 4), q = ((int)idx & 15) * 4;
        const int g = batch[i];
        const float4 av = *(const float4*)(act + (size_t)i * HF + q);
        float* p = pooled + g * HF + q;
        atomicAdd(p + 0, av.x);
        atomicAdd(p + 1, av.y);
        atomicAdd(p + 2, av.z);
        atomicAdd(p + 3, av.w);
        if (q == 0) atomicAdd(&cnt[g], 1.0f);
    }
}

__global__ void final_linear_kernel(const float* __restrict__ pooled,
                                    const float* __restrict__ cnt,
                                    const float* __restrict__ Wl,
                                    const float* __restrict__ bl,
                                    float* __restrict__ out) {
    int idx = threadIdx.x;                 // 640 = 64 graphs x 10 classes
    if (idx < NGRAPH * NCLASS) {
        const int g = idx / NCLASS, c = idx % NCLASS;
        const float inv = 1.0f / fmaxf(cnt[g], 1.0f);
        float acc = bl[c];
        #pragma unroll
        for (int k = 0; k < HF; ++k)
            acc += pooled[g * HF + k] * inv * Wl[k * NCLASS + c];
        out[idx] = acc;
    }
}

// ---------------- host side ----------------
static inline size_t align256(size_t x) { return (x + 255) & ~(size_t)255; }

extern "C" void kernel_launch(void* const* d_in, const int* in_sizes, int n_in,
                              void* d_out, int out_size, void* d_ws, size_t ws_size,
                              hipStream_t stream) {
    const float* x     = (const float*)d_in[0];
    const int*   ei    = (const int*)d_in[1];     // [2, E] flat: row0=src, row1=tgt
    const int*   batch = (const int*)d_in[2];
    const float* Wg[4] = { (const float*)d_in[3], (const float*)d_in[5],
                           (const float*)d_in[7], (const float*)d_in[9] };
    const float* bg[4] = { (const float*)d_in[4], (const float*)d_in[6],
                           (const float*)d_in[8], (const float*)d_in[10] };
    const float* Wl = (const float*)d_in[11];
    const float* bl = (const float*)d_in[12];
    float* out = (float*)d_out;

    const int n = in_sizes[0] / HF;
    const int e = in_sizes[1] / 2;
    const int* src = ei;
    const int* tgt = ei + e;

    // workspace layout (bufA has 1 pad row for the GEMM dump-store)
    char* ws = (char*)d_ws;
    size_t off = 0;
    float* dis  = (float*)(ws + off); off += align256((size_t)n * sizeof(float));
    float* bufA = (float*)(ws + off); off += align256((size_t)(n + 1) * HF * sizeof(float));
    float* bufB = (float*)(ws + off); off += align256((size_t)n * HF * sizeof(float));
    float* pooled = (float*)(ws + off); off += align256((size_t)NGRAPH * HF * sizeof(float));
    float* cnt    = (float*)(ws + off);
    (void)ws_size; (void)n_in; (void)out_size;

    const int B = 256;
    deg_init_kernel<<<(n + B - 1) / B, B, 0, stream>>>(dis, n);
    deg_edges_kernel<<<(e + B - 1) / B, B, 0, stream>>>(tgt, dis, e);
    deg_rsqrt_kernel<<<(n + B - 1) / B, B, 0, stream>>>(dis, n);

    const int gemmGrid = (n + 127) / 128;          // 8 waves x 16 rows per block
    const long long n16 = (long long)n * 16;
    const long long e16 = (long long)e * 16;
    const int n16Grid = (int)((n16 + B - 1) / B);
    const int e16Grid = (int)((e16 + B - 1) / B);

    // 4 GCN layers (ReLU of previous layer folded into GEMM A-load)
    for (int layer = 0; layer < 4; ++layer) {
        const float* actIn = (layer == 0) ? x : bufB;
        const int relu = (layer == 0) ? 0 : 1;
        gcn_gemm_wmma_kernel<<<gemmGrid, B, 0, stream>>>(actIn, Wg[layer], bufA, n, relu);
        agg_init_kernel<<<n16Grid, B, 0, stream>>>(bufA, dis, bg[layer], bufB, n);
        agg_edges_kernel<<<e16Grid, B, 0, stream>>>(src, tgt, bufA, dis, bufB, e);
    }

    // global mean pool + classifier (layer-4 output has no ReLU)
    pool_zero_kernel<<<(NGRAPH * HF + B - 1) / B, B, 0, stream>>>(pooled, cnt);
    pool_sum_kernel<<<n16Grid, B, 0, stream>>>(bufB, batch, pooled, cnt, n);
    final_linear_kernel<<<1, 640, 0, stream>>>(pooled, cnt, Wl, bl, out);
}
